// MemTransformerLM_81260781241041
// MI455X (gfx1250) — matrix operand order repaired and checked by reading, not compiled, not run on hardware
//
#include <hip/hip_runtime.h>

// ---------------------------------------------------------------------------
// Problem constants (Transformer-XL layer)
// ---------------------------------------------------------------------------
#define QLEN   1024
#define MLEN   1024
#define KLEN   2048
#define NHEAD  16
#define DHEAD  64
#define BSZ    2
#define DMODEL 1024
#define DINNER 4096

typedef __attribute__((ext_vector_type(16))) __bf16 v16bf;
typedef __attribute__((ext_vector_type(8)))  float  v8f;

union Frag16 {
    v16bf v;
    unsigned int u[8];
};

__device__ __forceinline__ float bf2f(unsigned short u) {
    union { unsigned int i; float f; } c;
    c.i = ((unsigned int)u) << 16;
    return c.f;
}

// hardware bf16 conversion (RNE) -- lets backend emit native cvt ops
__device__ __forceinline__ unsigned short f2bf(float f) {
    return __builtin_bit_cast(unsigned short, (__bf16)f);
}

__device__ __forceinline__ unsigned int pack_addbias(unsigned int u, float blo, float bhi) {
    float lo = bf2f((unsigned short)(u & 0xffffu)) + blo;
    float hi = bf2f((unsigned short)(u >> 16)) + bhi;
    return (unsigned int)f2bf(lo) | ((unsigned int)f2bf(hi) << 16);
}

__device__ __forceinline__ int imin(int a, int b) { return a < b ? a : b; }

// ---------------------------------------------------------------------------
// f32 -> bf16 conversion
// ---------------------------------------------------------------------------
__global__ void k_f32_to_bf16(const float* __restrict__ src,
                              unsigned short* __restrict__ dst, int n) {
    int i = blockIdx.x * blockDim.x + threadIdx.x;
    if (i < n) dst[i] = f2bf(src[i]);
}

// ---------------------------------------------------------------------------
// Generic bf16 WMMA GEMM:  C[M,N] = A[M,K] * B[K,N]  (+bias, relu, resid)
// WG = 256 threads = 8 waves. Block tile 64x128, K step 32.
// Wave (w>>1) -> 16-row band, (w&1) -> 64-col band; 4 WMMA accumulators
// sharing one A fragment per K step (wmma:ds_load ratio 4:10).
// ---------------------------------------------------------------------------
template <bool BIAS, bool RELU, bool RESID, bool OUTF, bool OUTB>
__global__ __launch_bounds__(256) void k_gemm_bf16(
    const unsigned short* __restrict__ A,
    const unsigned short* __restrict__ B,
    const float* __restrict__ bias,
    const float* __restrict__ resid,
    float* __restrict__ outF,
    unsigned short* __restrict__ outB,
    int M, int N, int K)
{
    __shared__ __align__(16) unsigned short As[64 * 32];   // [row][k]
    __shared__ __align__(16) unsigned short Bt[128 * 32];  // [n][k] (transposed)

    const int tid  = threadIdx.x;
    const int lane = tid & 31;
    const int wave = tid >> 5;
    const int half = lane >> 4;
    const int ln   = lane & 15;
    const int wm   = (wave >> 1) * 16;   // 4 row bands of 16
    const int wc   = (wave & 1) * 64;    // 2 col bands of 64
    const int m0   = blockIdx.y * 64;
    const int n0   = blockIdx.x * 128;

    // loader coordinates
    const int arow = tid >> 2;           // 64 rows, 8 bf16 per thread
    const int acol = (tid & 3) * 8;
    const int brow = tid >> 3;           // 32 k-rows, 16 bf16 per thread
    const int bcol = (tid & 7) * 16;

    v8f acc[4] = {v8f{}, v8f{}, v8f{}, v8f{}};

    for (int kt = 0; kt < K; kt += 32) {
        // ---- stage A tile (row-major) ----
        const uint4 av = *(const uint4*)(A + (size_t)(m0 + arow) * K + kt + acol);
        ((uint4*)As)[tid] = av;

        // ---- stage B tile transposed: Bt[n][k] ----
        union { uint4 v; unsigned short s[8]; } bv0, bv1;
        const unsigned short* bp = B + (size_t)(kt + brow) * N + n0 + bcol;
        bv0.v = *(const uint4*)bp;
        bv1.v = *(const uint4*)(bp + 8);
#pragma unroll
        for (int j = 0; j < 8; ++j) {
            Bt[(bcol + j) * 32 + brow]     = bv0.s[j];
            Bt[(bcol + 8 + j) * 32 + brow] = bv1.s[j];
        }

        if (kt + 32 < K) {
            __builtin_prefetch(A + (size_t)(m0 + arow) * K + kt + 32 + acol, 0, 1);
            __builtin_prefetch(B + (size_t)(kt + 32 + brow) * N + n0 + bcol, 0, 1);
        }
        __syncthreads();

        // ---- A fragment: m = ln (within wave band), K pairs per layout ----
        Frag16 a;
#pragma unroll
        for (int v = 0; v < 8; ++v) {
            const int kc = half * 16 + 2 * v;
            a.u[v] = *(const unsigned int*)&As[(wm + ln) * 32 + kc];
        }
        // ---- 4 B fragments (16-col tiles), 4 WMMAs reuse A ----
#pragma unroll
        for (int f = 0; f < 4; ++f) {
            Frag16 b;
#pragma unroll
            for (int v = 0; v < 8; ++v) {
                const int kc = half * 16 + 2 * v;
                b.u[v] = *(const unsigned int*)&Bt[(wc + f * 16 + ln) * 32 + kc];
            }
            acc[f] = __builtin_amdgcn_wmma_f32_16x16x32_bf16(false, a.v, false, b.v,
                                                             (short)0, acc[f], false, false);
        }
        __syncthreads();
    }

    // ---- epilogue ----
#pragma unroll
    for (int f = 0; f < 4; ++f) {
#pragma unroll
        for (int v = 0; v < 8; ++v) {
            const int m = m0 + wm + v + half * 8;
            const int n = n0 + wc + f * 16 + ln;
            float x = acc[f][v];
            if (BIAS)  x += bias[n];
            if (RELU)  x = fmaxf(x, 0.f);
            if (RESID) x += resid[(size_t)m * N + n];
            if (OUTF)  outF[(size_t)m * N + n] = x;
            if (OUTB)  outB[(size_t)m * N + n] = f2bf(x);
        }
    }
}

// ---------------------------------------------------------------------------
// Fused flash attention with Transformer-XL relative shift.
// Grid: (QLEN/64, NHEAD, BSZ), block 128 (4 waves, 16 q-rows per wave).
// heads: bf16 [KLEN*BSZ, 3*DMODEL]  (q | k | v column blocks)
// rk   : bf16 [KLEN, NHEAD*DHEAD]
// vec  : bf16 [QLEN*BSZ, DMODEL]  output (softmax(score) @ V)
// BD[i,j] = qr_i . rk[j + QLEN-1 - i]  (rel_shift applied analytically)
// K tile staged with GLOBAL_LOAD_ASYNC_TO_LDS_B128 (ASYNCcnt-tracked DMA).
// ---------------------------------------------------------------------------
__global__ __launch_bounds__(128) void k_attention(
    const unsigned short* __restrict__ heads,
    const unsigned short* __restrict__ rk,
    const float* __restrict__ rwb,
    const float* __restrict__ rrb,
    unsigned short* __restrict__ vec)
{
    __shared__ __align__(16) unsigned short Ks[32 * 64];   // [key][d]
    __shared__ __align__(16) unsigned short Vt[64 * 32];   // [d][key] transposed
    __shared__ float          Pbd[4][16 * 48];             // per-wave BD pre-shift
    __shared__ unsigned short Pp[4][16 * 32];              // per-wave probabilities

    constexpr float SCALE = 0.125f;                 // 1/sqrt(64)
    constexpr float LOG2E = 1.4426950408889634f;

    const int tid  = threadIdx.x;
    const int lane = tid & 31;
    const int wave = tid >> 5;
    const int half = lane >> 4;
    const int ln   = lane & 15;
    const int i0wg = blockIdx.x * 64;
    const int i0   = i0wg + wave * 16;
    const int h    = blockIdx.y;
    const int bi   = blockIdx.z;

    // ---- build q fragments with r_w_bias / r_r_bias folded in ----
    Frag16 qw0, qw1, qr0, qr1;
    {
        const int qrow = (MLEN + i0 + ln) * BSZ + bi;
        const unsigned short* qp = heads + (size_t)qrow * (3 * DMODEL) + h * DHEAD;
#pragma unroll
        for (int v = 0; v < 8; ++v) {
            const int kc = half * 16 + 2 * v;
            const unsigned int u0 = *(const unsigned int*)(qp + kc);
            const unsigned int u1 = *(const unsigned int*)(qp + kc + 32);
            qw0.u[v] = pack_addbias(u0, rwb[h * 64 + kc],      rwb[h * 64 + kc + 1]);
            qw1.u[v] = pack_addbias(u1, rwb[h * 64 + kc + 32], rwb[h * 64 + kc + 33]);
            qr0.u[v] = pack_addbias(u0, rrb[h * 64 + kc],      rrb[h * 64 + kc + 1]);
            qr1.u[v] = pack_addbias(u1, rrb[h * 64 + kc + 32], rrb[h * 64 + kc + 33]);
        }
    }

    v8f O[4] = {v8f{}, v8f{}, v8f{}, v8f{}};
    float mrow[8], lrow[8];
#pragma unroll
    for (int v = 0; v < 8; ++v) { mrow[v] = -3.0e38f; lrow[v] = 0.f; }

    const int ntiles = ((i0wg + 63 + MLEN) >> 5) + 1;

    for (int t = 0; t < ntiles; ++t) {
        const int j0 = t * 32;

        // ---- cooperative K / V staging (32 keys x 64 d) ----
        {
            const int e   = tid * 16;            // 16 bf16 = 32B per lane
            const int key = e >> 6;
            const int d0  = e & 63;
            const int kg  = imin(j0 + key, KLEN - 1);

            // K tile: async global->LDS DMA, no VGPR round trip
            const unsigned short* kp =
                heads + (size_t)(kg * BSZ + bi) * (3 * DMODEL) + DMODEL + h * DHEAD + d0;
            const unsigned int       ldsa  = (unsigned int)(size_t)(&Ks[e]);
            const unsigned long long gaddr = (unsigned long long)(size_t)kp;
            asm volatile(
                "global_load_async_to_lds_b128 %0, %1, off\n\t"
                "global_load_async_to_lds_b128 %0, %1, off offset:16"
                :: "v"(ldsa), "v"(gaddr) : "memory");

            // V tile: register transpose into Vt[d][key]
            const unsigned short* vp =
                heads + (size_t)(kg * BSZ + bi) * (3 * DMODEL) + 2 * DMODEL + h * DHEAD + d0;
            union { uint4 v; unsigned short s[8]; } v0, v1;
            v0.v = *(const uint4*)vp;
            v1.v = *(const uint4*)(vp + 8);
#pragma unroll
            for (int j = 0; j < 8; ++j) {
                Vt[(d0 + j) * 32 + key]     = v0.s[j];
                Vt[(d0 + 8 + j) * 32 + key] = v1.s[j];
            }
            asm volatile("s_wait_asynccnt 0x0" ::: "memory");
        }
        __syncthreads();

        // ---- AC = qw . K^T : two 16x16 score tiles ----
        v8f s0 = {}, s1 = {};
        {
            Frag16 bk;
#pragma unroll
            for (int v = 0; v < 8; ++v)
                bk.u[v] = *(const unsigned int*)&Ks[ln * 64 + half * 16 + 2 * v];
            s0 = __builtin_amdgcn_wmma_f32_16x16x32_bf16(false, qw0.v, false, bk.v,
                                                         (short)0, s0, false, false);
#pragma unroll
            for (int v = 0; v < 8; ++v)
                bk.u[v] = *(const unsigned int*)&Ks[ln * 64 + 32 + half * 16 + 2 * v];
            s0 = __builtin_amdgcn_wmma_f32_16x16x32_bf16(false, qw1.v, false, bk.v,
                                                         (short)0, s0, false, false);
#pragma unroll
            for (int v = 0; v < 8; ++v)
                bk.u[v] = *(const unsigned int*)&Ks[(16 + ln) * 64 + half * 16 + 2 * v];
            s1 = __builtin_amdgcn_wmma_f32_16x16x32_bf16(false, qw0.v, false, bk.v,
                                                         (short)0, s1, false, false);
#pragma unroll
            for (int v = 0; v < 8; ++v)
                bk.u[v] = *(const unsigned int*)&Ks[(16 + ln) * 64 + 32 + half * 16 + 2 * v];
            s1 = __builtin_amdgcn_wmma_f32_16x16x32_bf16(false, qw1.v, false, bk.v,
                                                         (short)0, s1, false, false);
        }

        // ---- BD pre-shift block: cols c0..c0+47, dumped to per-wave LDS ----
        const int c0 = j0 + (QLEN - 1) - (i0 + 15);
#pragma unroll
        for (int ct = 0; ct < 3; ++ct) {
            const int rrow = imin(c0 + ct * 16 + ln, KLEN - 1);
            const unsigned short* rp = rk + (size_t)rrow * (NHEAD * DHEAD) + h * DHEAD;
            v8f p = {};
            Frag16 br;
#pragma unroll
            for (int v = 0; v < 8; ++v)
                br.u[v] = *(const unsigned int*)(rp + half * 16 + 2 * v);
            p = __builtin_amdgcn_wmma_f32_16x16x32_bf16(false, qr0.v, false, br.v,
                                                        (short)0, p, false, false);
#pragma unroll
            for (int v = 0; v < 8; ++v)
                br.u[v] = *(const unsigned int*)(rp + 32 + half * 16 + 2 * v);
            p = __builtin_amdgcn_wmma_f32_16x16x32_bf16(false, qr1.v, false, br.v,
                                                        (short)0, p, false, false);
#pragma unroll
            for (int v = 0; v < 8; ++v)
                Pbd[wave][(v + half * 8) * 48 + ct * 16 + ln] = p[v];
        }

        // ---- score = (AC + rel-shifted BD) * scale, causal+mem mask ----
#pragma unroll
        for (int v = 0; v < 8; ++v) {
            const int m = v + half * 8;
            const int i = i0 + m;
            const float bd0 = Pbd[wave][m * 48 + (ln + 15 - m)];
            const float bd1 = Pbd[wave][m * 48 + (16 + ln + 15 - m)];
            float x0 = (s0[v] + bd0) * SCALE;
            float x1 = (s1[v] + bd1) * SCALE;
            if (j0 + ln > i + MLEN)      x0 = -3.0e38f;
            if (j0 + 16 + ln > i + MLEN) x1 = -3.0e38f;
            s0[v] = x0;
            s1[v] = x1;
        }

        // ---- online softmax (row = 16 lanes within a half) ----
        float corr[8];
#pragma unroll
        for (int v = 0; v < 8; ++v) {
            float rm = fmaxf(s0[v], s1[v]);
            rm = fmaxf(rm, __shfl_xor(rm, 1, 32));
            rm = fmaxf(rm, __shfl_xor(rm, 2, 32));
            rm = fmaxf(rm, __shfl_xor(rm, 4, 32));
            rm = fmaxf(rm, __shfl_xor(rm, 8, 32));
            const float mnew = fmaxf(mrow[v], rm);
            corr[v] = exp2f((mrow[v] - mnew) * LOG2E);
            const float p0 = exp2f((s0[v] - mnew) * LOG2E);
            const float p1 = exp2f((s1[v] - mnew) * LOG2E);
            float rs = p0 + p1;
            rs += __shfl_xor(rs, 1, 32);
            rs += __shfl_xor(rs, 2, 32);
            rs += __shfl_xor(rs, 4, 32);
            rs += __shfl_xor(rs, 8, 32);
            lrow[v] = lrow[v] * corr[v] + rs;
            mrow[v] = mnew;
            const int m = v + half * 8;
            Pp[wave][m * 32 + ln]      = f2bf(p0);
            Pp[wave][m * 32 + 16 + ln] = f2bf(p1);
        }

        // rescale output accumulators
#pragma unroll
        for (int v = 0; v < 8; ++v) {
            O[0][v] *= corr[v];
            O[1][v] *= corr[v];
            O[2][v] *= corr[v];
            O[3][v] *= corr[v];
        }

        // ---- P @ V (contraction over 32 keys) ----
        Frag16 pa;
#pragma unroll
        for (int v = 0; v < 8; ++v) {
            const int kc = half * 16 + 2 * v;
            pa.u[v] = *(const unsigned int*)&Pp[wave][ln * 32 + kc];
        }
#pragma unroll
        for (int dt = 0; dt < 4; ++dt) {
            Frag16 bv;
#pragma unroll
            for (int v = 0; v < 8; ++v)
                bv.u[v] = *(const unsigned int*)&Vt[(dt * 16 + ln) * 32 + half * 16 + 2 * v];
            O[dt] = __builtin_amdgcn_wmma_f32_16x16x32_bf16(false, pa.v, false, bv.v,
                                                            (short)0, O[dt], false, false);
        }
        __syncthreads();
    }

    // ---- normalize and write vec (bf16) ----
#pragma unroll
    for (int v = 0; v < 8; ++v) {
        const float inv = 1.0f / lrow[v];
        const int i = i0 + v + half * 8;
        unsigned short* vp = vec + (size_t)(i * BSZ + bi) * DMODEL + h * DHEAD;
#pragma unroll
        for (int dt = 0; dt < 4; ++dt)
            vp[dt * 16 + ln] = f2bf(O[dt][v] * inv);
    }
}

// ---------------------------------------------------------------------------
// LayerNorm over rows of D=1024; optional bf16 twin output.
// ---------------------------------------------------------------------------
__global__ __launch_bounds__(256) void k_layernorm(
    const float* __restrict__ x, const float* __restrict__ g,
    const float* __restrict__ b, float* __restrict__ outF,
    unsigned short* __restrict__ outB, int D)
{
    __shared__ float2 red[256];
    const float* row = x + (size_t)blockIdx.x * D;
    float s = 0.f, s2 = 0.f;
    for (int i = threadIdx.x; i < D; i += 256) {
        const float v = row[i];
        s += v;
        s2 += v * v;
    }
    red[threadIdx.x] = make_float2(s, s2);
    __syncthreads();
    for (int st = 128; st > 0; st >>= 1) {
        if (threadIdx.x < st) {
            red[threadIdx.x].x += red[threadIdx.x + st].x;
            red[threadIdx.x].y += red[threadIdx.x + st].y;
        }
        __syncthreads();
    }
    const float mu  = red[0].x / (float)D;
    const float var = red[0].y / (float)D - mu * mu;
    const float inv = rsqrtf(var + 1e-5f);
    for (int i = threadIdx.x; i < D; i += 256) {
        const float v = (row[i] - mu) * inv * g[i] + b[i];
        outF[(size_t)blockIdx.x * D + i] = v;
        if (outB) outB[(size_t)blockIdx.x * D + i] = f2bf(v);
    }
}

// ---------------------------------------------------------------------------
// Host launcher
// ---------------------------------------------------------------------------
extern "C" void kernel_launch(void* const* d_in, const int* in_sizes, int n_in,
                              void* d_out, int out_size, void* d_ws, size_t ws_size,
                              hipStream_t stream) {
    (void)in_sizes; (void)n_in; (void)out_size; (void)ws_size;

    const float* w    = (const float*)d_in[0];
    const float* r    = (const float*)d_in[1];
    const float* mems = (const float*)d_in[2];
    // d_in[3] attn_mask: causal+memory mask computed analytically
    const float* W_qkv = (const float*)d_in[4];
    const float* W_r   = (const float*)d_in[5];
    const float* W_o   = (const float*)d_in[6];
    const float* rwb   = (const float*)d_in[7];
    const float* rrb   = (const float*)d_in[8];
    const float* ln1g  = (const float*)d_in[9];
    const float* ln1b  = (const float*)d_in[10];
    const float* ffW1  = (const float*)d_in[11];
    const float* ffb1  = (const float*)d_in[12];
    const float* ffW2  = (const float*)d_in[13];
    const float* ffb2  = (const float*)d_in[14];
    const float* ln2g  = (const float*)d_in[15];
    const float* ln2b  = (const float*)d_in[16];
    float* out = (float*)d_out;

    // ---- workspace carve-up ----
    char* ws = (char*)d_ws;
    size_t off = 0;
    auto take = [&](size_t bytes) -> char* {
        char* p = ws + off;
        off = (off + bytes + 255) & ~(size_t)255;
        return p;
    };
    unsigned short* catbf  = (unsigned short*)take((size_t)KLEN * BSZ * DMODEL * 2);
    unsigned short* rbf    = (unsigned short*)take((size_t)KLEN * DMODEL * 2);
    unsigned short* wqkvbf = (unsigned short*)take((size_t)DMODEL * 3 * DMODEL * 2);
    unsigned short* wrbf   = (unsigned short*)take((size_t)DMODEL * DMODEL * 2);
    unsigned short* wobf   = (unsigned short*)take((size_t)DMODEL * DMODEL * 2);
    unsigned short* w1bf   = (unsigned short*)take((size_t)DMODEL * DINNER * 2);
    unsigned short* w2bf   = (unsigned short*)take((size_t)DINNER * DMODEL * 2);
    unsigned short* headsbf= (unsigned short*)take((size_t)KLEN * BSZ * 3 * DMODEL * 2);
    unsigned short* rkbf   = (unsigned short*)take((size_t)KLEN * DMODEL * 2);
    unsigned short* vecbf  = (unsigned short*)take((size_t)QLEN * BSZ * DMODEL * 2);
    float*          sum1   = (float*)take((size_t)QLEN * BSZ * DMODEL * 4);
    float*          hf     = (float*)take((size_t)QLEN * BSZ * DMODEL * 4);
    unsigned short* hbf    = (unsigned short*)take((size_t)QLEN * BSZ * DMODEL * 2);
    unsigned short* ff1bf  = (unsigned short*)take((size_t)QLEN * BSZ * DINNER * 2);
    float*          sum2   = (float*)take((size_t)QLEN * BSZ * DMODEL * 4);

    auto cvt = [&](const float* s, unsigned short* d, int n) {
        k_f32_to_bf16<<<(n + 255) / 256, 256, 0, stream>>>(s, d, n);
    };

    // ---- 1. convert inputs / weights to bf16 ----
    const int nTok = MLEN * BSZ * DMODEL;          // mems block
    cvt(mems, catbf, nTok);                        // cat[:mlen]
    cvt(w, catbf + nTok, QLEN * BSZ * DMODEL);     // cat[mlen:]
    cvt(r, rbf, KLEN * DMODEL);
    cvt(W_qkv, wqkvbf, DMODEL * 3 * DMODEL);
    cvt(W_r, wrbf, DMODEL * DMODEL);
    cvt(W_o, wobf, DMODEL * DMODEL);
    cvt(ffW1, w1bf, DMODEL * DINNER);
    cvt(ffW2, w2bf, DINNER * DMODEL);

    const int ROWS = QLEN * BSZ; // 2048

    // ---- 2. heads = cat @ W_qkv  [4096 x 3072] ----
    k_gemm_bf16<false, false, false, false, true>
        <<<dim3(3 * DMODEL / 128, KLEN * BSZ / 64), 256, 0, stream>>>(
            catbf, wqkvbf, nullptr, nullptr, nullptr, headsbf,
            KLEN * BSZ, 3 * DMODEL, DMODEL);

    // ---- 3. rk = r @ W_r  [2048 x 1024] ----
    k_gemm_bf16<false, false, false, false, true>
        <<<dim3(DMODEL / 128, KLEN / 64), 256, 0, stream>>>(
            rbf, wrbf, nullptr, nullptr, nullptr, rkbf,
            KLEN, DMODEL, DMODEL);

    // ---- 4. fused flash attention (AC + rel-shift BD + softmax + PV) ----
    k_attention<<<dim3(QLEN / 64, NHEAD, BSZ), 128, 0, stream>>>(
        headsbf, rkbf, rwb, rrb, vecbf);

    // ---- 5. attn_out = vec @ W_o + w (residual) -> sum1 (f32) ----
    k_gemm_bf16<false, false, true, true, false>
        <<<dim3(DMODEL / 128, ROWS / 64), 256, 0, stream>>>(
            vecbf, wobf, nullptr, w, sum1, nullptr,
            ROWS, DMODEL, DMODEL);

    // ---- 6. h = LN1(sum1) -> hf (f32) + hbf (bf16) ----
    k_layernorm<<<ROWS, 256, 0, stream>>>(sum1, ln1g, ln1b, hf, hbf, DMODEL);

    // ---- 7. ff1 = relu(h @ W1 + b1) -> bf16 ----
    k_gemm_bf16<true, true, false, false, true>
        <<<dim3(DINNER / 128, ROWS / 64), 256, 0, stream>>>(
            hbf, w1bf, ffb1, nullptr, nullptr, ff1bf,
            ROWS, DINNER, DMODEL);

    // ---- 8. ff2 = ff1 @ W2 + b2 + h (residual) -> sum2 (f32) ----
    k_gemm_bf16<true, false, true, true, false>
        <<<dim3(DMODEL / 128, ROWS / 64), 256, 0, stream>>>(
            ff1bf, w2bf, ffb2, hf, sum2, nullptr,
            ROWS, DMODEL, DINNER);

    // ---- 9. out = LN2(sum2) ----
    k_layernorm<<<ROWS, 256, 0, stream>>>(sum2, ln2g, ln2b, out, nullptr, DMODEL);
}
